// IGFormer_9010841387607
// MI455X (gfx1250) — matrix operand order repaired and checked by reading, not compiled, not run on hardware
//
#include <hip/hip_runtime.h>

// ---------------------------------------------------------------------------
// IGFormer linear-attention pipeline for MI455X (gfx1250, wave32, WMMA, TDM).
//   K0 zero accumulators -> K1 global mean/var reduce -> K2 weight->fragment
//   swizzle -> K3 fused LN + QKV GEMM (bf16 WMMA), q row-major, k/v col-major
//   -> K5 kvs = K^T V per head (WMMA, direct global fragment loads) ->
//   K6 scale + kvs fragment swizzle ->
//   K7 out = (q @ kvs_s + N*v) / (q . kss_s + N)  (WMMA, q tile via TDM)
// ---------------------------------------------------------------------------

#define N_ROWS 65536
#define DIN    512
#define NOUT   1536           // q|k|v concatenated output columns
#define EPS_LN 1e-5f
#define LDS_STRIDE 520        // 512 + 8 bf16 pad -> bank-conflict-free b128 reads

typedef __attribute__((ext_vector_type(16))) __bf16 v16bf;
typedef __attribute__((ext_vector_type(8)))  __bf16 v8bf;
typedef __attribute__((ext_vector_type(4)))  __bf16 v4bf;
typedef __attribute__((ext_vector_type(8)))  float  v8f;
typedef __attribute__((ext_vector_type(4)))  unsigned int u32x4;
typedef __attribute__((ext_vector_type(8)))  int i32x8;
typedef __attribute__((ext_vector_type(4)))  int i32x4;

// workspace float-index offsets (float* base)
#define WSF_SUM    0          // [0]=sum, [1]=sumsq, [2]=|q|^2, [3]=|k|^2
#define WSF_BCAT   1024       // 1536 biases
#define WSF_KSSUM  4096       // 512
#define WSF_KSS_S  5120       // 512 scaled
#define WSF_KVS    8192       // 8*64*64 fp32
// workspace byte offsets for bf16 buffers
#define WSB_WFRAG  (262144ull)                 // 16*96*32*16 bf16 = 1.5 MB
#define WSB_KVSFRG (2097152ull)                // 8*2*4*32*16 bf16 = 64 KB
#define WSB_QBUF   (4194304ull)                // N*512 bf16, ROW-major
#define WSB_KBUF   (71303168ull)               // 512*N bf16, COL-major
#define WSB_VBUF   (138412032ull)              // 512*N bf16, COL-major

__device__ __forceinline__ v8f wmma_bf16(v16bf a, v16bf b, v8f c) {
  return __builtin_amdgcn_wmma_f32_16x16x32_bf16(false, a, false, b,
                                                 (short)0, c, false, false);
}

// A-fragment (16x32 bf16): rowptr points at k0 of this lane's row; K values
// are contiguous.  lanes<16 take K {0..15}, lanes>=16 take {16..31}.
__device__ __forceinline__ v16bf load_a_rows(const __bf16* rowptr, int lane) {
  const int off = (lane & 16) ? 16 : 0;
  v8bf lo = *(const v8bf*)(rowptr + off);
  v8bf hi = *(const v8bf*)(rowptr + off + 8);
  return __builtin_shufflevector(lo, hi, 0, 1, 2, 3, 4, 5, 6, 7,
                                 8, 9, 10, 11, 12, 13, 14, 15);
}

__device__ __forceinline__ float wave_reduce(float v) {
#pragma unroll
  for (int off = 16; off > 0; off >>= 1) v += __shfl_down(v, off, 32);
  return v;
}

// --------------------------- K0: zero accumulators -------------------------
__global__ void igf_zero(float* __restrict__ wsf) {
  int i = blockIdx.x * blockDim.x + threadIdx.x;
  if (i < 40960) wsf[i] = 0.0f;   // stats + bcat + kssum + kss_s + kvs
}

// --------------------------- K1: global sum / sumsq ------------------------
__global__ void igf_reduce(const float* __restrict__ h, float* __restrict__ wsf) {
  const size_t total4 = (size_t)N_ROWS * DIN / 4;
  const size_t stride = (size_t)gridDim.x * blockDim.x;
  float s1 = 0.f, s2 = 0.f;
  for (size_t i = (size_t)blockIdx.x * blockDim.x + threadIdx.x; i < total4;
       i += stride) {
    __builtin_prefetch(&((const float4*)h)[i + stride], 0, 0);
    float4 x = ((const float4*)h)[i];
    s1 += x.x + x.y + x.z + x.w;
    s2 += x.x * x.x + x.y * x.y + x.z * x.z + x.w * x.w;
  }
  s1 = wave_reduce(s1);
  s2 = wave_reduce(s2);
  __shared__ float ls[16];
  int lane = threadIdx.x & 31, wave = threadIdx.x >> 5;
  if (lane == 0) { ls[wave] = s1; ls[wave + 8] = s2; }
  __syncthreads();
  if (threadIdx.x == 0) {
    float a = 0.f, b = 0.f;
#pragma unroll
    for (int w = 0; w < 8; ++w) { a += ls[w]; b += ls[w + 8]; }
    atomicAdd(&wsf[WSF_SUM + 0], a);
    atomicAdd(&wsf[WSF_SUM + 1], b);
  }
}

// ----------------- K2: swizzle Wq|Wk|Wv into B-fragment layout -------------
// frag idx = ((ck*96+cn)*32+lane)*16+t ;  B[k][n] = W[n][k] (x @ W^T)
__global__ void igf_wprep(const float* __restrict__ wq, const float* __restrict__ wk,
                          const float* __restrict__ wv, const float* __restrict__ bq,
                          const float* __restrict__ bk, const float* __restrict__ bv,
                          __bf16* __restrict__ wfrag, float* __restrict__ wsf) {
  int idx = blockIdx.x * blockDim.x + threadIdx.x;     // [0, 786432)
  int t = idx & 15;
  int r = idx >> 4;
  int lane = r & 31;  r >>= 5;
  int cn = r % 96;
  int ck = r / 96;
  int k = ck * 32 + ((lane & 16) ? 16 + t : t);
  int n = cn * 16 + (lane & 15);
  float w = (n < 512)    ? wq[n * 512 + k]
            : (n < 1024) ? wk[(n - 512) * 512 + k]
                         : wv[(n - 1024) * 512 + k];
  wfrag[idx] = (__bf16)w;
  if (idx < NOUT) {
    float b = (idx < 512) ? bq[idx] : (idx < 1024) ? bk[idx - 512] : bv[idx - 1024];
    wsf[WSF_BCAT + idx] = b;
  }
}

// --------- K3: fused LayerNorm + QKV GEMM + norm/ks_sum reductions ---------
__global__ void igf_qkv(const float* __restrict__ h, float* __restrict__ wsf,
                        const __bf16* __restrict__ wfrag,
                        __bf16* __restrict__ qbuf, __bf16* __restrict__ kbuf,
                        __bf16* __restrict__ vbuf) {
  extern __shared__ char smem[];
  __bf16* xs = (__bf16*)smem;                                // 128 x 520 bf16
  float* ksl = (float*)(smem + 128 * LDS_STRIDE * 2);        // 512 f32

  const int row0 = blockIdx.x * 128;
  const int tid = threadIdx.x;
  const int lane = tid & 31, wave = tid >> 5;

  const float Mtot = (float)N_ROWS * (float)DIN;
  const float mu = wsf[0] / Mtot;
  const float inv = rsqrtf(fmaxf(wsf[1] / Mtot - mu * mu, 0.f) + EPS_LN);

  for (int i = tid; i < 512; i += 256) ksl[i] = 0.f;

  // stage normalized x tile (128x512) as bf16 in LDS
  for (int i = tid; i < 128 * 128; i += 256) {
    int r = i >> 7, c4 = i & 127;
    float4 x = *(const float4*)(h + (size_t)(row0 + r) * DIN + c4 * 4);
    v4bf o;
    o[0] = (__bf16)((x.x - mu) * inv);
    o[1] = (__bf16)((x.y - mu) * inv);
    o[2] = (__bf16)((x.z - mu) * inv);
    o[3] = (__bf16)((x.w - mu) * inv);
    *(v4bf*)(xs + r * LDS_STRIDE + c4 * 4) = o;
  }
  __syncthreads();

  const int mrow = wave * 16 + (lane & 15);          // this lane's LDS row
  const int grow0 = row0 + wave * 16 + ((lane & 16) ? 8 : 0);
  float qn2 = 0.f, kn2 = 0.f;

  for (int nc = 0; nc < NOUT; nc += 64) {
    v8f acc[4] = {};
    for (int ck = 0; ck < 16; ++ck) {
      v16bf a = load_a_rows(xs + mrow * LDS_STRIDE + ck * 32, lane);
      const __bf16* bb =
          wfrag + ((size_t)(ck * 96 + (nc >> 4)) * 32 + lane) * 16;
#pragma unroll
      for (int dt = 0; dt < 4; ++dt)
        acc[dt] = wmma_bf16(a, *(const v16bf*)(bb + (size_t)dt * 512), acc[dt]);
    }
    // epilogue: bias add, route to q (row-major) / k,v (col-major)
#pragma unroll
    for (int dt = 0; dt < 4; ++dt) {
      const int j0 = nc + dt * 16;
      const int j = j0 + (lane & 15);
      const float bias = wsf[WSF_BCAT + j];
      if (j0 < 512) {                                   // Q: row-major
#pragma unroll
        for (int r2 = 0; r2 < 8; ++r2) {
          float val = acc[dt][r2] + bias;
          qbuf[(size_t)(grow0 + r2) * DIN + j] = (__bf16)val;
          qn2 += val * val;
        }
      } else if (j0 < 1024) {                           // K: col-major packed
        v8bf pk;
        float kcol = 0.f;
#pragma unroll
        for (int r2 = 0; r2 < 8; ++r2) {
          float val = acc[dt][r2] + bias;
          pk[r2] = (__bf16)val;
          kn2 += val * val;
          kcol += val;
        }
        *(v8bf*)(kbuf + (size_t)(j - 512) * N_ROWS + grow0) = pk;
        atomicAdd(&ksl[j - 512], kcol);
      } else {                                          // V: col-major packed
        v8bf pv;
#pragma unroll
        for (int r2 = 0; r2 < 8; ++r2)
          pv[r2] = (__bf16)(acc[dt][r2] + bias);
        *(v8bf*)(vbuf + (size_t)(j - 1024) * N_ROWS + grow0) = pv;
      }
    }
  }

  qn2 = wave_reduce(qn2);
  kn2 = wave_reduce(kn2);
  if (lane == 0) {
    atomicAdd(&wsf[2], qn2);
    atomicAdd(&wsf[3], kn2);
  }
  __syncthreads();
  for (int i = tid; i < 512; i += 256) atomicAdd(&wsf[WSF_KSSUM + i], ksl[i]);
}

// ------------------- K5: kvs[h] += K_h^T V_h  (per-head WMMA) --------------
// k/v are col-major, so the transposed-A fragment of K is two contiguous b128
// loads per lane and the B fragment of V is one contiguous 32B load per lane.
__global__ void igf_kvs(const __bf16* __restrict__ kbuf,
                        const __bf16* __restrict__ vbuf,
                        float* __restrict__ wsf) {
  const int lane = threadIdx.x & 31, hhead = threadIdx.x >> 5;  // wave == head
  v8f acc[16] = {};

  for (int c = blockIdx.x; c < N_ROWS / 64; c += gridDim.x) {
    const int r0 = c * 64;
#pragma unroll
    for (int l0 = 0; l0 < 64; l0 += 32) {
      v16bf afr[4];
#pragma unroll
      for (int mt = 0; mt < 4; ++mt) {
        const int m = hhead * 64 + mt * 16 + (lane & 15);
        afr[mt] = load_a_rows(kbuf + (size_t)m * N_ROWS + r0 + l0, lane);
      }
#pragma unroll
      for (int dt = 0; dt < 4; ++dt) {
        const int d = hhead * 64 + dt * 16 + (lane & 15);
        v16bf bfr = *(const v16bf*)(vbuf + (size_t)d * N_ROWS + r0 + l0 +
                                    ((lane & 16) ? 16 : 0));
#pragma unroll
        for (int mt = 0; mt < 4; ++mt)
          acc[mt * 4 + dt] = wmma_bf16(afr[mt], bfr, acc[mt * 4 + dt]);
      }
    }
  }

  float* kvs = wsf + WSF_KVS;
#pragma unroll
  for (int mt = 0; mt < 4; ++mt)
#pragma unroll
    for (int dt = 0; dt < 4; ++dt)
#pragma unroll
      for (int r2 = 0; r2 < 8; ++r2) {
        int m = mt * 16 + ((lane & 16) ? 8 + r2 : r2);
        int d = dt * 16 + (lane & 15);
        atomicAdd(&kvs[(hhead * 64 + m) * 64 + d], acc[mt * 4 + dt][r2]);
      }
}

// ------- K6: fold 1/(|q||k|) into kvs & ks_sum; swizzle kvs fragments ------
__global__ void igf_scale(float* __restrict__ wsf, __bf16* __restrict__ kvsfrag) {
  int idx = blockIdx.x * blockDim.x + threadIdx.x;     // [0, 32768)
  float s = rsqrtf(wsf[2]) * rsqrtf(wsf[3]);
  if (idx < 512) wsf[WSF_KSS_S + idx] = wsf[WSF_KSSUM + idx] * s;
  int t = idx & 15;
  int r = idx >> 4;
  int lane = r & 31;  r >>= 5;
  int dt = r & 3;     r >>= 2;
  int ck = r & 1;
  int hh = r >> 1;
  int m = ck * 32 + ((lane & 16) ? 16 : 0) + t;
  int d = dt * 16 + (lane & 15);
  kvsfrag[idx] = (__bf16)(wsf[WSF_KVS + (hh * 64 + m) * 64 + d] * s);
}

// ----------- K7: out = (q @ kvs_s + N*v) / (q . kss_s + N) -----------------
__global__ void igf_final(const __bf16* __restrict__ qbuf,
                          const __bf16* __restrict__ vbuf,
                          const __bf16* __restrict__ kvsfrag,
                          const float* __restrict__ wsf,
                          float* __restrict__ out) {
  extern __shared__ char smem[];
  __bf16* qs = (__bf16*)smem;                          // 128 x 520
  const int row0 = blockIdx.x * 128;
  const int tid = threadIdx.x, lane = tid & 31, wave = tid >> 5;

  // Stage the q tile (128 x 512 bf16, LDS pitch 520) with the Tensor Data
  // Mover: 2D tile, data_size=2B, pad 4 DWORDs (16B) after every 256 DWORDs
  // (one 1024B row) -> LDS row stride 1040B = 520 bf16.  Wave 0 issues the
  // DMA (wave-uniform branch; other waves never see the instruction).
  if (wave == 0) {
    unsigned lds = (unsigned)(size_t)(void*)qs;
    unsigned long long ga =
        (unsigned long long)(size_t)(qbuf + (size_t)row0 * DIN);
    u32x4 g0;
    g0[0] = 1u;                                        // count=1, user mode
    g0[1] = lds;                                       // lds_addr [63:32]
    g0[2] = (unsigned)ga;                              // global_addr lo
    g0[3] = (unsigned)(ga >> 32) | 0x80000000u;        // addr hi | type=2
    i32x8 g1;
    g1[0] = (1 << 16) | (1 << 20) | (7 << 22) | (3 << 25);
    //        data=2B    pad_en     intvl=256dw  amount=4dw
    g1[1] = (int)(512u << 16);                         // tensor_dim0 = 512
    g1[2] = (int)(128u << 16);                         // tensor_dim1 = 128
    g1[3] = (int)(512u << 16);                         // tile_dim0  = 512
    g1[4] = 128;                                       // tile_dim1  = 128
    g1[5] = 512;                                       // tensor_dim0_stride
    g1[6] = 0;
    g1[7] = 0;
    i32x4 gz4 = {0, 0, 0, 0};
    i32x8 gz8 = {0, 0, 0, 0, 0, 0, 0, 0};
    __builtin_amdgcn_tensor_load_to_lds(g0, g1, gz4, gz4, gz8, 0);
    __builtin_amdgcn_s_wait_tensorcnt(0);
  }
  __syncthreads();

  const int mrow = wave * 16 + (lane & 15);
  const int grow0 = row0 + wave * 16 + ((lane & 16) ? 8 : 0);
  const float* kss = wsf + WSF_KSS_S;

  for (int hh = 0; hh < 8; ++hh) {
    v8f acc[4] = {};
#pragma unroll
    for (int ck = 0; ck < 2; ++ck) {
      v16bf a = load_a_rows(qs + mrow * LDS_STRIDE + hh * 64 + ck * 32, lane);
      const __bf16* bb =
          kvsfrag + ((size_t)((hh * 2 + ck) * 4) * 32 + lane) * 16;
#pragma unroll
      for (int dt = 0; dt < 4; ++dt)
        acc[dt] = wmma_bf16(a, *(const v16bf*)(bb + (size_t)dt * 512), acc[dt]);
    }
    // denominator for this lane's row (lanes>=16 compute a redundant copy)
    float den = 0.f;
    const __bf16* qrow = qs + mrow * LDS_STRIDE + hh * 64;
#pragma unroll
    for (int m = 0; m < 64; ++m) den += (float)qrow[m] * kss[hh * 64 + m];
    den += 65536.0f;

#pragma unroll
    for (int dt = 0; dt < 4; ++dt) {
      const int j = hh * 64 + dt * 16 + (lane & 15);
      v8bf vv = *(const v8bf*)(vbuf + (size_t)j * N_ROWS + grow0);
#pragma unroll
      for (int r2 = 0; r2 < 8; ++r2) {
        int mloc = (lane & 16) ? 8 + r2 : r2;
        float dh = __shfl(den, mloc, 32);
        float val = (acc[dt][r2] + 65536.0f * (float)vv[r2]) / dh;
        out[(size_t)(grow0 + r2) * DIN + j] = val;
      }
    }
  }
}

// ---------------------------------------------------------------------------
extern "C" void kernel_launch(void* const* d_in, const int* in_sizes, int n_in,
                              void* d_out, int out_size, void* d_ws, size_t ws_size,
                              hipStream_t stream) {
  (void)in_sizes; (void)n_in; (void)out_size; (void)ws_size;
  const float* h  = (const float*)d_in[0];
  const float* wq = (const float*)d_in[1];
  const float* bq = (const float*)d_in[2];
  const float* wk = (const float*)d_in[3];
  const float* bk = (const float*)d_in[4];
  const float* wv = (const float*)d_in[5];
  const float* bv = (const float*)d_in[6];
  float* out = (float*)d_out;

  char* ws = (char*)d_ws;
  float* wsf = (float*)ws;
  __bf16* wfrag   = (__bf16*)(ws + WSB_WFRAG);
  __bf16* kvsfrag = (__bf16*)(ws + WSB_KVSFRG);
  __bf16* qbuf    = (__bf16*)(ws + WSB_QBUF);
  __bf16* kbuf    = (__bf16*)(ws + WSB_KBUF);
  __bf16* vbuf    = (__bf16*)(ws + WSB_VBUF);

  const size_t smem_qkv = 128 * LDS_STRIDE * 2 + 512 * 4;  // x tile + ks partial
  const size_t smem_fin = 128 * LDS_STRIDE * 2;            // q tile

  igf_zero  <<<160, 256, 0, stream>>>(wsf);
  igf_reduce<<<2048, 256, 0, stream>>>(h, wsf);
  igf_wprep <<<3072, 256, 0, stream>>>(wq, wk, wv, bq, bk, bv, wfrag, wsf);
  igf_qkv   <<<N_ROWS / 128, 256, smem_qkv, stream>>>(h, wsf, wfrag,
                                                      qbuf, kbuf, vbuf);
  igf_kvs   <<<64, 256, 0, stream>>>(kbuf, vbuf, wsf);
  igf_scale <<<128, 256, 0, stream>>>(wsf, kvsfrag);
  igf_final <<<N_ROWS / 128, 256, smem_fin, stream>>>(qbuf, vbuf, kvsfrag,
                                                      wsf, out);
}
